// AttentionHead_Hybrid_67430986547734
// MI455X (gfx1250) — matrix operand order repaired and checked by reading, not compiled, not run on hardware
//
#include <hip/hip_runtime.h>
#include <hip/hip_bf16.h>

#define BATCH 32
#define SEQ   1024
#define DK    64
#define LEN_PHI 2016

typedef __attribute__((ext_vector_type(16))) _Float16 v16h;
typedef __attribute__((ext_vector_type(8)))  float    v8f;

union Frag16 { v16h h; unsigned int u[8]; };

// A-matrix (16-bit, 16xK) element index for VGPR v on this lane:
// lanes 0-15: V0..3 -> K={0,1..6,7}, V4..7 -> K={16,17..22,23}; lanes 16-31: +8
__device__ __forceinline__ int kmapA(int v, int lane) {
    return 2 * v + ((v >= 4) ? 8 : 0) + ((lane & 16) ? 8 : 0);
}
// B-matrix (16-bit, Kx16): same K order as A (lanes 0-15: K=0..7,16..23; lanes 16-31: +8)
__device__ __forceinline__ int kmapB(int v, int lane) {
    return 2 * v + ((v >= 4) ? 8 : 0) + ((lane & 16) ? 8 : 0);
}

// ---- CDNA5 async global->LDS copies (ASYNCcnt); inline asm = toolchain-portable
__device__ __forceinline__ void async_copy_b128(const void* g, void* l) {
    unsigned lds = (unsigned)(size_t)l;                 // aperture: low 32b = LDS offset
    unsigned long long ga = (unsigned long long)(size_t)g;
    asm volatile("global_load_async_to_lds_b128 %0, %1, off"
                 :: "v"(lds), "v"(ga) : "memory");
}
__device__ __forceinline__ void async_copy_b32(const void* g, void* l) {
    unsigned lds = (unsigned)(size_t)l;
    unsigned long long ga = (unsigned long long)(size_t)g;
    asm volatile("global_load_async_to_lds_b32 %0, %1, off"
                 :: "v"(lds), "v"(ga) : "memory");
}
__device__ __forceinline__ void wait_async0() {
    asm volatile("s_wait_asynccnt 0" ::: "memory");
}

// ---------------------------------------------------------------------------
// Kernel 1: O = product of 2016 Givens (RBS) rotations applied to I.
// Column-parallel: thread e owns column e entirely -> no synchronization.
// ---------------------------------------------------------------------------
__global__ void build_orthogonal_kernel(const float* __restrict__ phi,
                                        float* __restrict__ O) {
    __shared__ float Ol[DK * DK];   // [row d][col e]
    const int e = threadIdx.x;      // 64 threads
    for (int d = 0; d < DK; ++d) Ol[d * DK + e] = (d == e) ? 1.0f : 0.0f;
    int i = 0, j = 1;
    for (int s = 0; s < LEN_PHI; ++s) {
        float th = phi[s];          // scalar broadcast load
        float sn, c;
        sincosf(th, &sn, &c);
        float ri = Ol[i * DK + e];
        float rj = Ol[j * DK + e];
        Ol[i * DK + e] = c * ri + sn * rj;
        Ol[j * DK + e] = -sn * ri + c * rj;
        if (++j == DK) { ++i; j = i + 1; }
    }
    for (int d = 0; d < DK; ++d) O[d * DK + e] = Ol[d * DK + e];
}

// ---------------------------------------------------------------------------
// Kernel 2: per-row norms, xh (f16), W = xh @ O (f16),
//           V^T = (x @ Vw^T + Vb)^T stored [b][d][n] so kernel-3 staging is
//           contiguous 16B lane-copies for the async DMA path.
// ---------------------------------------------------------------------------
__global__ __launch_bounds__(256) void precompute_kernel(
    const float* __restrict__ x, const float* __restrict__ O,
    const float* __restrict__ Vw, const float* __restrict__ Vb,
    float* __restrict__ norms, _Float16* __restrict__ xh,
    _Float16* __restrict__ W, _Float16* __restrict__ Vt_ws) {
    __shared__ float Ol[DK * DK];          // [d][e] -> conflict-free (bank = e)
    __shared__ float Vwl[DK * (DK + 1)];   // [e][d] padded to 65 -> bank = (e+d)%64
    __shared__ float xrow[4][DK];
    const int tid = threadIdx.x;
    for (int idx = tid; idx < DK * DK; idx += 256) Ol[idx] = O[idx];
    for (int idx = tid; idx < DK * DK; idx += 256) {
        int e = idx >> 6, d = idx & 63;
        Vwl[e * (DK + 1) + d] = Vw[idx];
    }
    __syncthreads();
    const int g = tid >> 6;   // row group
    const int e = tid & 63;   // output column
    const float vb = Vb[e];
    const long long row0 = (long long)blockIdx.x * 64;
    for (int it = 0; it < 16; ++it) {
        const long long row = row0 + it * 4 + g;
        xrow[g][e] = x[row * DK + e];
        __syncthreads();
        float ssq = 0.f, accO = 0.f, accV = 0.f;
        for (int d = 0; d < DK; ++d) {
            float v = xrow[g][d];                 // broadcast within group
            ssq  += v * v;
            accO += v * Ol[d * DK + e];
            accV += v * Vwl[e * (DK + 1) + d];
        }
        float nrm = sqrtf(ssq);
        float inv = 1.0f / nrm;
        if (e == 0) norms[row] = nrm;
        xh[row * DK + e] = (_Float16)(xrow[g][e] * inv);
        W[row * DK + e]  = (_Float16)(accO * inv);
        const long long bb = row >> 10, n = row & (SEQ - 1);
        Vt_ws[(bb * DK + e) * SEQ + n] = (_Float16)(accV + vb);
        __syncthreads();
    }
}

// ---------------------------------------------------------------------------
// Kernel 3: fused flash-style attention + residual + LayerNorm.
// 256 threads (8 wave32s); wave w owns a 16-row query tile.
// Double-buffered key/value staging via GLOBAL_LOAD_ASYNC_TO_LDS (ASYNCcnt)
// overlapped with 8 v_wmma_f32_16x16x32_f16 per 32-key chunk.
// ---------------------------------------------------------------------------
__global__ __launch_bounds__(256) void attention_kernel(
    const float* __restrict__ x, const _Float16* __restrict__ xh,
    const _Float16* __restrict__ Wq, const _Float16* __restrict__ Vt_ws,
    const float* __restrict__ norms, const float* __restrict__ gamma,
    const float* __restrict__ beta, float* __restrict__ out) {
    __shared__ __align__(16) _Float16 xhm[2][32][72];   // key rows [m][k]; 144B stride
    __shared__ __align__(16) _Float16 Vt[2][64][40];    // V^T [d][m]; 80B stride
    __shared__ __align__(16) _Float16 Pl[8][16][40];    // per-wave P tile [q][m]
    __shared__ float normm[2][32];

    const int tid  = threadIdx.x;
    const int wave = tid >> 5;
    const int lane = tid & 31;
    const int col  = lane & 15;
    const int half8 = (lane & 16) ? 8 : 0;
    const int b = blockIdx.y;
    const int qbase = blockIdx.x * 128 + wave * 16;
    const size_t brow = (size_t)b * SEQ;

    // issue async DMA of all chunk data: each thread copies 16B of xh tile,
    // 16B of V^T tile; first 32 threads copy one norm each.
    auto stage = [&](int buf, int chunk) {
        const int mbase = chunk * 32;
        {   // xh chunk: 32 rows x 128B contiguous
            int r = tid >> 3, seg = tid & 7;
            const char* g = (const char*)(xh + (brow + mbase) * DK) + r * 128 + seg * 16;
            async_copy_b128(g, &xhm[buf][r][seg * 8]);
        }
        {   // V^T chunk: 64 rows x 64B (row-contiguous thanks to transposed ws)
            int d = tid >> 2, seg = tid & 3;
            const _Float16* g = Vt_ws + ((size_t)b * DK + d) * SEQ + mbase + seg * 8;
            async_copy_b128(g, &Vt[buf][d][seg * 8]);
        }
        if (tid < 32) async_copy_b32(norms + brow + mbase + tid, &normm[buf][tid]);
    };

    stage(0, 0);   // prologue DMA overlaps the fragment/state setup below

    // Loop-invariant A fragments: W query tile, K=0..31 and K=32..63
    Frag16 AW0, AW1;
    {
        const _Float16* wr = Wq + (brow + qbase + col) * DK;
        for (int v = 0; v < 8; ++v) {
            int k = kmapA(v, lane);
            AW0.u[v] = *(const unsigned int*)(wr + k);
            AW1.u[v] = *(const unsigned int*)(wr + k + 32);
        }
    }
    float nq[8];
    for (int r = 0; r < 8; ++r) nq[r] = norms[brow + qbase + r + half8];

    v8f acc[4];
    for (int t = 0; t < 4; ++t)
        for (int r = 0; r < 8; ++r) acc[t][r] = 0.f;
    float rowsum[8];
    for (int r = 0; r < 8; ++r) rowsum[r] = 0.f;

    wait_async0();
    __syncthreads();

    for (int chunk = 0; chunk < SEQ / 32; ++chunk) {
        const int cur = chunk & 1;
        if (chunk + 1 < SEQ / 32) stage(cur ^ 1, chunk + 1);  // overlap DMA w/ WMMA

        // -------- S = W_q @ xh_m^T (two 16x16 tiles, K=64), exp, stage P -----
        for (int mt = 0; mt < 2; ++mt) {
            Frag16 B0, B1;
            const int mrow = mt * 16 + col;
            for (int v = 0; v < 8; ++v) {
                int k = kmapB(v, lane);
                B0.u[v] = *(const unsigned int*)(&xhm[cur][mrow][k]);
                B1.u[v] = *(const unsigned int*)(&xhm[cur][mrow][k + 32]);
            }
            v8f c;
            for (int r = 0; r < 8; ++r) c[r] = 0.f;
            c = __builtin_amdgcn_wmma_f32_16x16x32_f16(false, AW0.h, false, B0.h,
                                                       (short)0, c, false, false);
            c = __builtin_amdgcn_wmma_f32_16x16x32_f16(false, AW1.h, false, B1.h,
                                                       (short)0, c, false, false);
            float nj = normm[cur][mt * 16 + col];
            for (int r = 0; r < 8; ++r) {
                float s = c[r];
                // logits bounded in [0, ~20] -> exp safe in f32 without max-sub
                float eprob = __expf(s * s * nq[r] * nj * 0.125f);
                rowsum[r] += eprob;
                Pl[wave][r + half8][mt * 16 + col] = (_Float16)eprob;
            }
        }

        // -------- acc += P(16x32) @ V(32x64) : 4 WMMAs --------
        Frag16 AP;
        for (int v = 0; v < 8; ++v) {
            int k = kmapA(v, lane);
            AP.u[v] = *(const unsigned int*)(&Pl[wave][col][k]);
        }
        for (int t = 0; t < 4; ++t) {
            Frag16 BV;
            for (int v = 0; v < 8; ++v) {
                int k = kmapB(v, lane);
                BV.u[v] = *(const unsigned int*)(&Vt[cur][t * 16 + col][k]);
            }
            acc[t] = __builtin_amdgcn_wmma_f32_16x16x32_f16(false, AP.h, false, BV.h,
                                                            (short)0, acc[t], false, false);
        }

        wait_async0();     // own async DMA for next buffer complete
        __syncthreads();   // all waves: done reading cur, next buffer visible
    }

    // -------- softmax normalize + residual + LayerNorm + store --------
    float inv_l[8];
    for (int r = 0; r < 8; ++r) {
        float s = rowsum[r];
        for (int off = 1; off < 16; off <<= 1) s += __shfl_xor(s, off, 32);
        inv_l[r] = 1.0f / s;
    }
    float g_ln[4], b_ln[4];
    for (int t = 0; t < 4; ++t) {
        g_ln[t] = gamma[t * 16 + col];
        b_ln[t] = beta[t * 16 + col];
    }
    for (int r = 0; r < 8; ++r) {
        const size_t q = brow + qbase + r + half8;
        float o[4];
        float ssum = 0.f, ssq = 0.f;
        for (int t = 0; t < 4; ++t) {
            float v = acc[t][r] * inv_l[r] + x[q * DK + t * 16 + col];
            o[t] = v;
            ssum += v;
            ssq += v * v;
        }
        for (int off = 1; off < 16; off <<= 1) {
            ssum += __shfl_xor(ssum, off, 32);
            ssq  += __shfl_xor(ssq,  off, 32);
        }
        float mu   = ssum * (1.0f / DK);
        float var  = ssq * (1.0f / DK) - mu * mu;
        float rstd = rsqrtf(var + 1e-5f);
        for (int t = 0; t < 4; ++t)
            out[q * DK + t * 16 + col] = (o[t] - mu) * rstd * g_ln[t] + b_ln[t];
    }
}

// ---------------------------------------------------------------------------
extern "C" void kernel_launch(void* const* d_in, const int* in_sizes, int n_in,
                              void* d_out, int out_size, void* d_ws, size_t ws_size,
                              hipStream_t stream) {
    const float* x     = (const float*)d_in[0];
    const float* Vw    = (const float*)d_in[1];
    const float* Vb    = (const float*)d_in[2];
    const float* phi   = (const float*)d_in[3];
    const float* gamma = (const float*)d_in[4];
    const float* beta  = (const float*)d_in[5];
    float* out = (float*)d_out;

    char* ws = (char*)d_ws;
    const size_t ROWS = (size_t)BATCH * SEQ;
    float*    O     = (float*)(ws);                                   // 16 KB
    float*    norms = (float*)(ws + 16384);                           // 128 KB
    _Float16* xh    = (_Float16*)(ws + 16384 + 131072);               // 4 MB
    _Float16* W     = (_Float16*)(ws + 16384 + 131072 + ROWS * DK * 2);
    _Float16* Vt    = (_Float16*)(ws + 16384 + 131072 + 2 * ROWS * DK * 2);

    build_orthogonal_kernel<<<1, 64, 0, stream>>>(phi, O);
    precompute_kernel<<<512, 256, 0, stream>>>(x, O, Vw, Vb, norms, xh, W, Vt);
    dim3 grid(SEQ / 128, BATCH);
    attention_kernel<<<grid, 256, 0, stream>>>(x, xh, W, Vt, norms, gamma, beta, out);

    (void)in_sizes; (void)n_in; (void)out_size; (void)ws_size;
}
